// SPPGN_76742475644968
// MI455X (gfx1250) — matrix operand order repaired and checked by reading, not compiled, not run on hardware
//
#include <hip/hip_runtime.h>

typedef __attribute__((ext_vector_type(16))) __bf16 v16bf;
typedef __attribute__((ext_vector_type(8)))  float  v8f;
typedef __attribute__((ext_vector_type(4)))  unsigned int v4u;
typedef __attribute__((ext_vector_type(8)))  int    v8i;
typedef __attribute__((ext_vector_type(4)))  int    v4i;

#define HD 128
#define BNC 0.9999950000374997f   /* 1/sqrt(1+1e-5) */
#define SQEPS 1e-12f

enum { F_BN = 1, F_BF16 = 2, F_RES = 4 };

__device__ __forceinline__ float ssqrt(float x) {
    return sqrtf(fmaxf(x, 0.f) + SQEPS) - sqrtf(fmaxf(-x, 0.f) + SQEPS);
}

// ---------- init: pair_h = pair_x @ W_pair   [P,16] x [16,128] ----------
__global__ void k_init(const float* __restrict__ px, const float* __restrict__ W,
                       float* __restrict__ ph, long total) {
    long t = (long)blockIdx.x * blockDim.x + threadIdx.x;
    if (t >= total) return;
    long p = t >> 7; int n = (int)(t & 127);
    float a = 0.f;
    #pragma unroll
    for (int k = 0; k < 16; ++k) a += px[p * 16 + k] * W[k * HD + n];
    ph[t] = a;
}

__global__ void k_add_edge(const float* __restrict__ eh, const int* __restrict__ epos,
                           float* __restrict__ ph, long total) {
    long t = (long)blockIdx.x * blockDim.x + threadIdx.x;
    if (t >= total) return;
    long e = t >> 7; int n = (int)(t & 127);
    atomicAdd(&ph[(long)epos[e] * HD + n], eh[t]);
}

__global__ void k_add_diag(const float* __restrict__ nh, const float* __restrict__ dx,
                           const float* __restrict__ W, const int* __restrict__ dpos,
                           float* __restrict__ ph, long total) {
    long t = (long)blockIdx.x * blockDim.x + threadIdx.x;
    if (t >= total) return;
    long i = t >> 7; int n = (int)(t & 127);
    float v = nh[t];
    #pragma unroll
    for (int k = 0; k < 16; ++k) v += dx[i * 16 + k] * W[k * HD + n];
    ph[(long)dpos[i] * HD + n] += v;   // diag positions are unique
}

// ---------- pack weights [K,128] f32 row-major -> B-fragment-swizzled bf16 ----------
// gemm reads Bswz[(((kc*8+nt)*32+lane)*16)+e]; fragment element e of lane maps to
// k = kc*32 + (e&7) + 8*(lane>=16) + 16*(e>=8) ; n = nt*16 + (lane&15)
__global__ void k_packB(const float* __restrict__ Wm, __bf16* __restrict__ dst, long total) {
    long t = (long)blockIdx.x * blockDim.x + threadIdx.x;
    if (t >= total) return;
    int e = (int)(t & 15), lane = (int)((t >> 4) & 31);
    int nt = (int)((t >> 9) & 7); long kc = t >> 12;
    int n = nt * 16 + (lane & 15);
    long k = kc * 32 + (e & 7) + 8 * (lane >> 4) + 16 * ((e >> 3) & 1);
    dst[t] = (__bf16)Wm[k * HD + n];
}

// ---------- pack activations -> A-fragment-swizzled bf16 ----------
// gemm reads Aswz[(((rt*Kch+kc)*32+lane)*16)+e]; element e of lane maps to
// m = rt*16 + (lane&15); k = kc*32 + (e&7) + 8*(lane>=16) + 16*(e>=8)
__global__ void k_packA_f32(const float* __restrict__ src, __bf16* __restrict__ dst,
                            long total, int Kch) {
    long t = (long)blockIdx.x * blockDim.x + threadIdx.x;
    if (t >= total) return;
    int e = (int)(t & 15), lane = (int)((t >> 4) & 31);
    long q = t >> 9; int kc = (int)(q % Kch); long rt = q / Kch;
    long m = rt * 16 + (lane & 15);
    int k = kc * 32 + (e & 7) + 8 * (lane >> 4) + 16 * ((e >> 3) & 1);
    dst[t] = (__bf16)src[m * (long)(Kch * 32) + k];
}

__global__ void k_packA_bf16(const __bf16* __restrict__ src, __bf16* __restrict__ dst,
                             long total, int Kch) {
    long t = (long)blockIdx.x * blockDim.x + threadIdx.x;
    if (t >= total) return;
    int e = (int)(t & 15), lane = (int)((t >> 4) & 31);
    long q = t >> 9; int kc = (int)(q % Kch); long rt = q / Kch;
    long m = rt * 16 + (lane & 15);
    int k = kc * 32 + (e & 7) + 8 * (lane >> 4) + 16 * ((e >> 3) & 1);
    dst[t] = src[m * (long)(Kch * 32) + k];
}

// concat([pair_h, signed_sqrt(agg)]) -> K=256 A-swizzle
__global__ void k_packA_cat(const float* __restrict__ ph, const float* __restrict__ agg,
                            __bf16* __restrict__ dst, long total) {
    long t = (long)blockIdx.x * blockDim.x + threadIdx.x;
    if (t >= total) return;
    int e = (int)(t & 15), lane = (int)((t >> 4) & 31);
    long q = t >> 9; int kc = (int)(q & 7); long rt = q >> 3;
    long m = rt * 16 + (lane & 15);
    int k = kc * 32 + (e & 7) + 8 * (lane >> 4) + 16 * ((e >> 3) & 1);
    float v = (k < HD) ? ph[m * HD + k] : ssqrt(agg[m * HD + (k - HD)]);
    dst[t] = (__bf16)v;
}

// ---------- WMMA GEMM: [M,K]bf16 x [K,128]bf16 -> [M,128], fused epilogues ----------
// B (shared by all 8 waves) is DMA'd once per block into LDS by the Tensor Data
// Mover; the K-loop then reads B fragments via ds_load while A streams through the
// global path. Each wave owns TWO 16-row tiles: 16 WMMAs per K-chunk.
__global__ __launch_bounds__(256) void k_gemm(
    const __bf16* __restrict__ A, const __bf16* __restrict__ B,
    const float* __restrict__ bias, const float* __restrict__ bng,
    const float* __restrict__ bnb, const float* __restrict__ res,
    float* __restrict__ outF, __bf16* __restrict__ outB,
    long M, int Kch, int flags) {
    extern __shared__ __bf16 Bs[];                      // Kch*4096 bf16 (32/64 KB)
    const int lane = threadIdx.x & 31;
    const int wave = threadIdx.x >> 5;

    if (threadIdx.x < 32) {                             // wave 0 issues the TDM DMA
        unsigned nd = (unsigned)Kch * 2048u;            // B size in dwords (<= 16384)
        unsigned ldsOff = (unsigned)(uintptr_t)(&Bs[0]);
        unsigned long long ga = (unsigned long long)(uintptr_t)B;
        v4u g0; v8i g1; v4i z4;
        g0[0] = 1u;                                     // count=1 (valid descriptor)
        g0[1] = ldsOff;                                 // lds_addr
        g0[2] = (unsigned)ga;                           // global_addr[31:0]
        g0[3] = (unsigned)((ga >> 32) & 0x1FFFFFFull) | (2u << 30);  // addr[56:32] | type=2
        g1[0] = 0x00020000;                             // wg_mask=0, data_size=4B
        g1[1] = (int)((nd & 0xFFFFu) << 16);            // tensor_dim0[15:0]
        g1[2] = (int)((nd >> 16) | (1u << 16));         // tensor_dim0[31:16], tensor_dim1=1
        g1[3] = (int)(nd << 16);                        // tile_dim0 = nd
        g1[4] = 1;                                      // tile_dim1 = 1
        g1[5] = (int)nd;                                // tensor_dim0_stride
        g1[6] = 0; g1[7] = 0;
        z4[0] = z4[1] = z4[2] = z4[3] = 0;
#if defined(__clang_major__) && (__clang_major__ >= 23)
        v8i z8; z8[0]=z8[1]=z8[2]=z8[3]=z8[4]=z8[5]=z8[6]=z8[7]=0;
        __builtin_amdgcn_tensor_load_to_lds(g0, g1, z4, z4, z8, 0);
#else
        __builtin_amdgcn_tensor_load_to_lds(g0, g1, z4, z4, 0);
#endif
        __builtin_amdgcn_s_wait_tensorcnt(0);
    }
    __syncthreads();                                    // B tile visible to all waves

    const long rt0 = (long)blockIdx.x * 16 + wave * 2;  // two row tiles per wave
    if (rt0 * 16 >= M) return;                          // (after barrier: safe)

    v8f acc0[8], acc1[8];
    #pragma unroll
    for (int i = 0; i < 8; ++i) {
        acc0[i] = (v8f){0.f, 0.f, 0.f, 0.f, 0.f, 0.f, 0.f, 0.f};
        acc1[i] = (v8f){0.f, 0.f, 0.f, 0.f, 0.f, 0.f, 0.f, 0.f};
    }

    const __bf16* ap0 = A + ((rt0 * Kch) * 32 + lane) * 16;          // 32B-aligned
    const __bf16* ap1 = A + (((rt0 + 1) * Kch) * 32 + lane) * 16;
    for (int kc = 0; kc < Kch; ++kc) {
        v16bf a0 = *(const v16bf*)ap0;  ap0 += 512;
        v16bf a1 = *(const v16bf*)ap1;  ap1 += 512;
        if (kc + 1 < Kch) {
            __builtin_prefetch(ap0, 0, 1);                            // global_prefetch_b8
            __builtin_prefetch(ap1, 0, 1);
        }
        const __bf16* bp = Bs + (((long)kc * 8) * 32 + lane) * 16;    // LDS fragments
        v16bf b[8];
        #pragma unroll
        for (int nt = 0; nt < 8; ++nt)
            b[nt] = *(const v16bf*)(bp + (long)nt * 512);             // ds_load_b128 x2
        #pragma unroll
        for (int nt = 0; nt < 8; ++nt) {
            acc0[nt] = __builtin_amdgcn_wmma_f32_16x16x32_bf16(
                false, a0, false, b[nt], (short)0, acc0[nt], false, false);
            acc1[nt] = __builtin_amdgcn_wmma_f32_16x16x32_bf16(
                false, a1, false, b[nt], (short)0, acc1[nt], false, false);
        }
    }

    const int g = lane >> 4, nl = lane & 15;
    #pragma unroll
    for (int nt = 0; nt < 8; ++nt) {
        int n = nt * 16 + nl;
        float bi = bias[n];
        float sc = 1.f, sh = 0.f;
        if (flags & F_BN) { sc = bng[n] * BNC; sh = bnb[n]; }
        #pragma unroll
        for (int r = 0; r < 8; ++r) {
            long m0 = rt0 * 16 + r + 8 * g;                           // C-layout: m=r+8*(lane>=16)
            long m1 = m0 + 16;
            float v0 = acc0[nt][r] + bi;
            float v1 = acc1[nt][r] + bi;
            if (flags & F_BN) {
                v0 = fmaxf(v0 * sc + sh, 0.f);                        // BN(eval) + ReLU
                v1 = fmaxf(v1 * sc + sh, 0.f);
            }
            if (flags & F_RES) {
                v0 += res[m0 * HD + n];                               // residual
                v1 += res[m1 * HD + n];
            }
            if (flags & F_BF16) {
                outB[m0 * HD + n] = (__bf16)v0;
                outB[m1 * HD + n] = (__bf16)v1;
            } else {
                outF[m0 * HD + n] = v0;
                outF[m1 * HD + n] = v1;
            }
        }
    }
}

// ---------- triple gather-multiply-scatter ----------
__global__ void k_zero(float* __restrict__ p, long total) {
    long t = (long)blockIdx.x * blockDim.x + threadIdx.x;
    if (t < total) p[t] = 0.f;
}

__global__ void k_triple(const __bf16* __restrict__ h1, const __bf16* __restrict__ h2,
                         const int* __restrict__ i0, const int* __restrict__ i1,
                         const int* __restrict__ i2, float* __restrict__ agg) {
    int t = blockIdx.x, n = threadIdx.x;
    float v = (float)h1[(long)i1[t] * HD + n] * (float)h2[(long)i2[t] * HD + n];
    atomicAdd(&agg[(long)i0[t] * HD + n], v);
}

// ---------- pooling + decode ----------
__global__ void k_pool(const float* __restrict__ ph, const int* __restrict__ dpos,
                       const int* __restrict__ ptr, const int* __restrict__ pxptr,
                       float* __restrict__ gh) {
    int g = blockIdx.x, n = threadIdx.x;
    int p0 = pxptr[g], p1 = pxptr[g + 1];
    float sAll = 0.f;
    for (int p = p0; p < p1; ++p) sAll += ph[(long)p * HD + n];
    int n0 = ptr[g], n1 = ptr[g + 1];
    float sD = 0.f;
    for (int i = n0; i < n1; ++i) sD += ph[(long)dpos[i] * HD + n];
    float nodes = (float)(n1 - n0), pairs = (float)(p1 - p0);
    gh[(long)g * 2 * HD + n]      = sD / nodes;
    gh[(long)g * 2 * HD + HD + n] = (sAll - sD) / fmaxf(pairs - nodes, 1.f);
}

__global__ void k_decode(const float* __restrict__ gh, const float* __restrict__ Wd,
                         const float* __restrict__ bd, float* __restrict__ out, int tasks) {
    __shared__ float red[256];
    int g = blockIdx.x, j = threadIdx.x;
    for (int task = 0; task < tasks; ++task) {
        red[j] = gh[(long)g * 256 + j] * Wd[(long)j * tasks + task];
        __syncthreads();
        for (int s = 128; s > 0; s >>= 1) {
            if (j < s) red[j] += red[j + s];
            __syncthreads();
        }
        if (j == 0) out[(long)g * tasks + task] = red[0] + bd[task];
        __syncthreads();
    }
}

static inline unsigned nb(long n, int bs) { return (unsigned)((n + bs - 1) / bs); }

extern "C" void kernel_launch(void* const* d_in, const int* in_sizes, int n_in,
                              void* d_out, int out_size, void* d_ws, size_t ws_size,
                              hipStream_t stream) {
    const float* pair_x = (const float*)d_in[0];
    const float* diag_x = (const float*)d_in[1];
    const float* node_h = (const float*)d_in[2];
    const float* edge_h = (const float*)d_in[3];
    const float* W_pair = (const float*)d_in[4];
    const float* W_diag = (const float*)d_in[5];
    const float* Wd     = (const float*)d_in[6];
    const float* bd     = (const float*)d_in[7];
    const float* m1_Wa = (const float*)d_in[8],  *m1_ba = (const float*)d_in[9];
    const float* m1_g  = (const float*)d_in[10], *m1_b  = (const float*)d_in[11];
    const float* m1_Wb = (const float*)d_in[12], *m1_bb = (const float*)d_in[13];
    const float* m2_Wa = (const float*)d_in[14], *m2_ba = (const float*)d_in[15];
    const float* m2_g  = (const float*)d_in[16], *m2_b  = (const float*)d_in[17];
    const float* m2_Wb = (const float*)d_in[18], *m2_bb = (const float*)d_in[19];
    const float* up_Wa = (const float*)d_in[20], *up_ba = (const float*)d_in[21];
    const float* up_g  = (const float*)d_in[22], *up_b  = (const float*)d_in[23];
    const float* up_Wb = (const float*)d_in[24], *up_bb = (const float*)d_in[25];
    const int* edge_pos = (const int*)d_in[26];
    const int* diag_pos = (const int*)d_in[27];
    const int* tri      = (const int*)d_in[28];
    const int* ptr      = (const int*)d_in[31];
    const int* pxptr    = (const int*)d_in[32];

    const long P = in_sizes[0] / 16;
    const long N = in_sizes[1] / 16;
    const long E = in_sizes[3] / HD;
    const int  G = in_sizes[31] - 1;
    const long T = in_sizes[28] / 3;
    const int  L = in_sizes[9] / HD;
    const int  TASKS = in_sizes[7];

    // workspace layout
    char* w = (char*)d_ws;
    float*  pair_h = (float*)w;  w += P * HD * 4;   // 128MB
    __bf16* pairA  = (__bf16*)w; w += P * HD * 2;   // A-swizzle of pair_h (K=128)
    __bf16* hidRM  = (__bf16*)w; w += P * HD * 2;   // hidden, row-major bf16
    __bf16* hidA   = (__bf16*)w; w += P * HD * 2;   // hidden, A-swizzle
    __bf16* h1     = (__bf16*)w; w += P * HD * 2;
    __bf16* h2     = (__bf16*)w; w += P * HD * 2;
    float*  aggb   = (float*)w;  w += P * HD * 4;   // 128MB
    __bf16* upA    = h1;                            // alias h1+h2 (P*256 bf16), dead by then
    __bf16* BsA1 = (__bf16*)w; w += 128 * HD * 2;
    __bf16* BsB1 = (__bf16*)w; w += 128 * HD * 2;
    __bf16* BsA2 = (__bf16*)w; w += 128 * HD * 2;
    __bf16* BsB2 = (__bf16*)w; w += 128 * HD * 2;
    __bf16* BsUa = (__bf16*)w; w += 256 * HD * 2;
    __bf16* BsUb = (__bf16*)w; w += 128 * HD * 2;
    float*  gh   = (float*)w;  w += (long)G * 2 * HD * 4;

    const int BS = 256;
    const unsigned gG = nb(P, 256);     // one block = 8 waves x 2 tiles = 256 rows
    const size_t lds128 = 128 * HD * 2; // 32KB B tile in LDS (K=128)
    const size_t lds256 = 256 * HD * 2; // 64KB B tile in LDS (K=256)

    k_init<<<nb(P * HD, BS), BS, 0, stream>>>(pair_x, W_pair, pair_h, P * HD);
    k_add_edge<<<nb(E * HD, BS), BS, 0, stream>>>(edge_h, edge_pos, pair_h, E * HD);
    k_add_diag<<<nb(N * HD, BS), BS, 0, stream>>>(node_h, diag_x, W_diag, diag_pos, pair_h, N * HD);

    for (int l = 0; l < L; ++l) {
        k_packB<<<nb(128 * HD, BS), BS, 0, stream>>>(m1_Wa + (long)l * 128 * HD, BsA1, 128 * HD);
        k_packB<<<nb(128 * HD, BS), BS, 0, stream>>>(m1_Wb + (long)l * 128 * HD, BsB1, 128 * HD);
        k_packB<<<nb(128 * HD, BS), BS, 0, stream>>>(m2_Wa + (long)l * 128 * HD, BsA2, 128 * HD);
        k_packB<<<nb(128 * HD, BS), BS, 0, stream>>>(m2_Wb + (long)l * 128 * HD, BsB2, 128 * HD);
        k_packB<<<nb(256 * HD, BS), BS, 0, stream>>>(up_Wa + (long)l * 256 * HD, BsUa, 256 * HD);
        k_packB<<<nb(128 * HD, BS), BS, 0, stream>>>(up_Wb + (long)l * 128 * HD, BsUb, 128 * HD);

        k_packA_f32<<<nb(P * HD, BS), BS, 0, stream>>>(pair_h, pairA, P * HD, 4);

        // m1 MLP
        k_gemm<<<gG, 256, lds128, stream>>>(pairA, BsA1, m1_ba + l * HD, m1_g + l * HD, m1_b + l * HD,
                                            nullptr, nullptr, hidRM, P, 4, F_BN | F_BF16);
        k_packA_bf16<<<nb(P * HD, BS), BS, 0, stream>>>(hidRM, hidA, P * HD, 4);
        k_gemm<<<gG, 256, lds128, stream>>>(hidA, BsB1, m1_bb + l * HD, nullptr, nullptr,
                                            nullptr, nullptr, h1, P, 4, F_BF16);
        // m2 MLP
        k_gemm<<<gG, 256, lds128, stream>>>(pairA, BsA2, m2_ba + l * HD, m2_g + l * HD, m2_b + l * HD,
                                            nullptr, nullptr, hidRM, P, 4, F_BN | F_BF16);
        k_packA_bf16<<<nb(P * HD, BS), BS, 0, stream>>>(hidRM, hidA, P * HD, 4);
        k_gemm<<<gG, 256, lds128, stream>>>(hidA, BsB2, m2_bb + l * HD, nullptr, nullptr,
                                            nullptr, nullptr, h2, P, 4, F_BF16);
        // triple gather-multiply-scatter
        k_zero<<<nb(P * HD, BS), BS, 0, stream>>>(aggb, P * HD);
        k_triple<<<(unsigned)T, HD, 0, stream>>>(h1, h2, tri, tri + T, tri + 2 * T, aggb);
        // up MLP (K=256 concat) + residual
        k_packA_cat<<<nb(P * HD * 2, BS), BS, 0, stream>>>(pair_h, aggb, upA, P * HD * 2);
        k_gemm<<<gG, 256, lds256, stream>>>(upA, BsUa, up_ba + l * HD, up_g + l * HD, up_b + l * HD,
                                            nullptr, nullptr, hidRM, P, 8, F_BN | F_BF16);
        k_packA_bf16<<<nb(P * HD, BS), BS, 0, stream>>>(hidRM, hidA, P * HD, 4);
        k_gemm<<<gG, 256, lds128, stream>>>(hidA, BsUb, up_bb + l * HD, nullptr, nullptr,
                                            pair_h, pair_h, nullptr, P, 4, F_RES);
    }

    k_pool<<<(unsigned)G, HD, 0, stream>>>(pair_h, diag_pos, ptr, pxptr, gh);
    k_decode<<<(unsigned)G, 256, 0, stream>>>(gh, Wd, bd, (float*)d_out, TASKS);
}
